// TFTDeepSeekAttention_73735998537979
// MI455X (gfx1250) — matrix operand order repaired and checked by reading, not compiled, not run on hardware
//
#include <hip/hip_runtime.h>
#include <hip/hip_bf16.h>
#include <math.h>

// ---------------------------------------------------------------------------
// Types for CDNA5 WMMA (wave32, 16x16x32 bf16 -> f32 accum)
// ---------------------------------------------------------------------------
typedef __attribute__((ext_vector_type(16))) __bf16 v16bf;
typedef __attribute__((ext_vector_type(8)))  __bf16 v8bf;
typedef __attribute__((ext_vector_type(8)))  float  v8f;

typedef unsigned short u16;
typedef unsigned int   u32;

#define SEQ   2048
#define DIM   2048
#define NHEAD 16
#define HD    128
#define RANK  512
#define NTOK  (2 * SEQ)          // B*S = 4096

static __device__ inline u16 f2bf(float f) {
  u32 u = __float_as_uint(f);
  u32 r = (u + 0x7FFFu + ((u >> 16) & 1u)) >> 16;   // round-to-nearest-even
  return (u16)r;
}

static __device__ inline v8f zero8() {
  v8f r;
#pragma unroll
  for (int i = 0; i < 8; ++i) r[i] = 0.0f;
  return r;
}

// Build a 16-element bf16 fragment from two 16-byte (8 x bf16) chunks.
static __device__ inline v16bf ld_frag(const u16* p0, const u16* p1) {
  v8bf a = *(const v8bf*)p0;
  v8bf b = *(const v8bf*)p1;
  v16bf r;
#pragma unroll
  for (int i = 0; i < 8; ++i) { r[i] = a[i]; r[8 + i] = b[i]; }
  return r;
}

static __device__ inline v8f wmma_bf16(v16bf a, v16bf b, v8f c) {
  return __builtin_amdgcn_wmma_f32_16x16x32_bf16(false, a, false, b,
                                                 (short)0, c, false, false);
}

// ---------------------------------------------------------------------------
// f32 -> bf16 conversion
// ---------------------------------------------------------------------------
__global__ void __launch_bounds__(256)
cvt_f32_to_bf16(const float* __restrict__ src, u16* __restrict__ dst, int n) {
  int i = blockIdx.x * blockDim.x + threadIdx.x;
  if (i < n) dst[i] = f2bf(src[i]);
}

// ---------------------------------------------------------------------------
// V = kron(S, I) applied to xt; written TRANSPOSED as vT[b][h][d][s] (bf16)
// so PV contraction (over keys s) is contiguous in memory.
// ---------------------------------------------------------------------------
__global__ void __launch_bounds__(256)
vmix_kernel(const float* __restrict__ xt, const float* __restrict__ kron,
            u16* __restrict__ vT) {
  int idx = blockIdx.x * blockDim.x + threadIdx.x;  // (((b*16+i)*128)+d)*2048+s
  if (idx >= 2 * NHEAD * HD * SEQ) return;
  int s = idx & (SEQ - 1);
  int t = idx >> 11;
  int d = t & (HD - 1); t >>= 7;
  int i = t & (NHEAD - 1);
  int b = t >> 4;
  const float* xrow = xt + ((size_t)(b * SEQ + s)) * DIM + d;
  float acc = 0.0f;
#pragma unroll
  for (int j = 0; j < NHEAD; ++j) acc += kron[i * NHEAD + j] * xrow[j * HD];
  vT[idx] = f2bf(acc);
}

// ---------------------------------------------------------------------------
// RMSNorm over last dim (RANK=512), one wave per token, bf16 output
// ---------------------------------------------------------------------------
__global__ void __launch_bounds__(128)
rmsnorm_kernel(const float* __restrict__ x, const float* __restrict__ w,
               u16* __restrict__ out, int ntok) {
  int wave = (int)((blockIdx.x * blockDim.x + threadIdx.x) >> 5);
  int lane = threadIdx.x & 31;
  if (wave >= ntok) return;
  const float* row = x + (size_t)wave * RANK;
  float ss = 0.0f;
  for (int i = lane; i < RANK; i += 32) { float v = row[i]; ss += v * v; }
#pragma unroll
  for (int m = 16; m >= 1; m >>= 1) ss += __shfl_xor(ss, m, 32);
  float inv = rsqrtf(ss * (1.0f / RANK) + 1e-6f);
  for (int i = lane; i < RANK; i += 32)
    out[(size_t)wave * RANK + i] = f2bf(row[i] * inv * w[i]);
}

// ---------------------------------------------------------------------------
// Generic GEMM: C[M,N] = A[M,K] * W[N,K]^T   (A, W bf16 row-major)
// One wave computes a 32x64 tile; K stepped by 32 -> 8 WMMAs per step.
// C_F32 = 1 -> f32 output, else bf16 output.
// ---------------------------------------------------------------------------
template <int C_F32>
__global__ void __launch_bounds__(32, 1)
gemm_bf16_wmma(const u16* __restrict__ A, const u16* __restrict__ W,
               void* __restrict__ Cp, int M, int N, int K) {
  int ntiles = N >> 6;
  int tile = blockIdx.x;
  int tm = tile / ntiles;
  int tn = tile % ntiles;
  int lane = threadIdx.x & 31;
  int l15 = lane & 15;
  int hi  = lane >> 4;
  int m0 = tm * 32, n0 = tn * 64;

  v8f acc[2][4];
#pragma unroll
  for (int mt = 0; mt < 2; ++mt)
#pragma unroll
    for (int t = 0; t < 4; ++t) acc[mt][t] = zero8();

  const u16* arow0 = A + (size_t)(m0 + l15) * K;
  const u16* arow1 = A + (size_t)(m0 + 16 + l15) * K;
  for (int k0 = 0; k0 < K; k0 += 32) {
    int abase = k0 + hi * 8;                       // ISA A-layout
    v16bf af0 = ld_frag(arow0 + abase, arow0 + abase + 16);
    v16bf af1 = ld_frag(arow1 + abase, arow1 + abase + 16);
    int bbase = k0 + hi * 8;                       // ISA B-layout
#pragma unroll
    for (int t = 0; t < 4; ++t) {
      const u16* wrow = W + (size_t)(n0 + t * 16 + l15) * K + bbase;
      v16bf bfrag = ld_frag(wrow, wrow + 16);
      acc[0][t] = wmma_bf16(af0, bfrag, acc[0][t]);
      acc[1][t] = wmma_bf16(af1, bfrag, acc[1][t]);
    }
  }

#pragma unroll
  for (int mt = 0; mt < 2; ++mt) {
    int mb = m0 + mt * 16;
    if (C_F32) {
      float* C = (float*)Cp;
#pragma unroll
      for (int t = 0; t < 4; ++t)
#pragma unroll
        for (int r = 0; r < 8; ++r)
          C[(size_t)(mb + r + 8 * hi) * N + n0 + t * 16 + l15] = acc[mt][t][r];
    } else {
      u16* C = (u16*)Cp;
#pragma unroll
      for (int t = 0; t < 4; ++t)
#pragma unroll
        for (int r = 0; r < 8; ++r)
          C[(size_t)(mb + r + 8 * hi) * N + n0 + t * 16 + l15] = f2bf(acc[mt][t][r]);
    }
  }
}

// ---------------------------------------------------------------------------
// Flash attention with causal ALiBi. One wave per (b, h, 16-query block).
// q,k: [B,S,H,HD] bf16; vT: [B,H,HD,S] bf16; attn out: [B,S,H*HD] bf16.
// ---------------------------------------------------------------------------
__global__ void __launch_bounds__(32, 1)
attn_kernel(const u16* __restrict__ q, const u16* __restrict__ k,
            const u16* __restrict__ vT, u16* __restrict__ attn) {
  int blk = blockIdx.x;                 // b*H*(S/16) + h*(S/16) + qb
  int qb = blk & 127; blk >>= 7;
  int h  = blk & (NHEAD - 1);
  int b  = blk >> 4;
  int q0 = qb * 16;
  int lane = threadIdx.x & 31;
  int l15 = lane & 15, hi = lane >> 4;

  __shared__ __align__(16) u16 pbuf[16 * 32];

  // Q fragments (A-layout), head dim 128 = 4 chunks of K=32
  v16bf qa[4];
  const u16* qrow = q + ((size_t)((b * SEQ + q0 + l15) * NHEAD + h)) * HD;
#pragma unroll
  for (int kc = 0; kc < 4; ++kc) {
    int base = kc * 32 + hi * 16;
    qa[kc] = ld_frag(qrow + base, qrow + base + 8);
  }

  const float slope = exp2f(-0.5f * (float)(h + 1));  // ALiBi, H=16 pow2
  const float scale = 0.088388347648318447f;          // 1/sqrt(128)

  float m_[8], l_[8];
  v8f acc[8];
#pragma unroll
  for (int r = 0; r < 8; ++r) { m_[r] = -INFINITY; l_[r] = 0.0f; }
#pragma unroll
  for (int t = 0; t < 8; ++t) acc[t] = zero8();

  const int jend = q0 + 16;                     // exclusive causal key limit
  for (int j0 = 0; j0 < jend; j0 += 32) {
    // ---- scores: two 16x16 tiles (keys j0..15, j0+16..31) ----
    v8f s0 = zero8(), s1 = zero8();
#pragma unroll
    for (int kc = 0; kc < 4; ++kc) {
      int bbase = kc * 32 + hi * 16;
      const u16* kr0 = k + ((size_t)((b * SEQ + j0 + l15) * NHEAD + h)) * HD + bbase;
      const u16* kr1 = k + ((size_t)((b * SEQ + j0 + 16 + l15) * NHEAD + h)) * HD + bbase;
      v16bf kb0 = ld_frag(kr0, kr0 + 8);
      v16bf kb1 = ld_frag(kr1, kr1 + 8);
      s0 = wmma_bf16(qa[kc], kb0, s0);
      s1 = wmma_bf16(qa[kc], kb1, s1);
    }

    // ---- scale + ALiBi + causal mask, online softmax ----
    int key0 = j0 + l15, key1 = j0 + 16 + l15;
    float p0[8], p1[8];
#pragma unroll
    for (int r = 0; r < 8; ++r) {
      int qr = q0 + r + 8 * hi;
      float v0 = (key0 <= qr) ? s0[r] * scale + slope * (float)(key0 - qr) : -INFINITY;
      float v1 = (key1 <= qr) ? s1[r] * scale + slope * (float)(key1 - qr) : -INFINITY;
      float mx = fmaxf(v0, v1);
#pragma unroll
      for (int sh = 1; sh < 16; sh <<= 1) mx = fmaxf(mx, __shfl_xor(mx, sh, 32));
      float mnew = fmaxf(m_[r], mx);
      float corr = expf(m_[r] - mnew);          // 0 when m_ was -inf
      float e0 = expf(v0 - mnew);
      float e1 = expf(v1 - mnew);
      float rs = e0 + e1;
#pragma unroll
      for (int sh = 1; sh < 16; sh <<= 1) rs += __shfl_xor(rs, sh, 32);
      l_[r] = l_[r] * corr + rs;
      m_[r] = mnew;
      p0[r] = e0; p1[r] = e1;
#pragma unroll
      for (int t = 0; t < 8; ++t) acc[t][r] *= corr;
    }

    // ---- repack P (C-layout) -> A-layout via LDS ----
#pragma unroll
    for (int r = 0; r < 8; ++r) {
      pbuf[(r + 8 * hi) * 32 + l15]      = f2bf(p0[r]);
      pbuf[(r + 8 * hi) * 32 + 16 + l15] = f2bf(p1[r]);
    }
    __syncthreads();
    const u16* prow = &pbuf[l15 * 32 + hi * 16];
    v16bf pa = ld_frag(prow, prow + 8);
    __syncthreads();

    // ---- PV: contraction over 32 keys, contiguous in vT ----
#pragma unroll
    for (int t = 0; t < 8; ++t) {
      const u16* vrow = vT + ((size_t)((b * NHEAD + h) * HD + t * 16 + l15)) * SEQ
                        + j0 + hi * 16;
      v16bf vb = ld_frag(vrow, vrow + 8);
      acc[t] = wmma_bf16(pa, vb, acc[t]);
    }
  }

  // ---- normalize and write [B,S,H*HD] bf16 ----
#pragma unroll
  for (int t = 0; t < 8; ++t)
#pragma unroll
    for (int r = 0; r < 8; ++r) {
      int row = q0 + r + 8 * hi;
      attn[((size_t)(b * SEQ + row)) * DIM + h * HD + t * 16 + l15] =
          f2bf(acc[t][r] / l_[r]);
    }
}

// ---------------------------------------------------------------------------
// Host-side orchestration
// ---------------------------------------------------------------------------
extern "C" void kernel_launch(void* const* d_in, const int* in_sizes, int n_in,
                              void* d_out, int out_size, void* d_ws, size_t ws_size,
                              hipStream_t stream) {
  (void)in_sizes; (void)n_in; (void)out_size; (void)ws_size;

  const float* x    = (const float*)d_in[0];   // [B,S,D]
  const float* xt   = (const float*)d_in[1];   // [B,S,D]
  const float* wq   = (const float*)d_in[2];   // [2048,2048]
  const float* wka  = (const float*)d_in[3];   // [512,2048]
  const float* knw  = (const float*)d_in[4];   // [512]
  const float* wkb  = (const float*)d_in[5];   // [2048,512]
  const float* kron = (const float*)d_in[6];   // [16,16]
  const float* wo   = (const float*)d_in[7];   // [2048,2048]

  char* ws = (char*)d_ws;
  size_t off = 0;
  auto alloc = [&](size_t bytes) {
    void* p = ws + off;
    off = (off + bytes + 255) & ~(size_t)255;
    return p;
  };

  u16* x_bf   = (u16*)alloc((size_t)NTOK * DIM * 2);
  u16* wq_bf  = (u16*)alloc((size_t)DIM * DIM * 2);
  u16* wka_bf = (u16*)alloc((size_t)RANK * DIM * 2);
  u16* wkb_bf = (u16*)alloc((size_t)DIM * RANK * 2);
  u16* wo_bf  = (u16*)alloc((size_t)DIM * DIM * 2);
  u16* q_bf   = (u16*)alloc((size_t)NTOK * DIM * 2);
  float* kc_f = (float*)alloc((size_t)NTOK * RANK * 4);
  u16* kc_bf  = (u16*)alloc((size_t)NTOK * RANK * 2);
  u16* k_bf   = (u16*)alloc((size_t)NTOK * DIM * 2);
  u16* vT_bf  = (u16*)alloc((size_t)NTOK * DIM * 2);
  u16* a_bf   = (u16*)alloc((size_t)NTOK * DIM * 2);

  // --- bf16 conversions ---
  {
    int n = NTOK * DIM;
    cvt_f32_to_bf16<<<(n + 255) / 256, 256, 0, stream>>>(x, x_bf, n);
  }
  {
    int n = DIM * DIM;
    cvt_f32_to_bf16<<<(n + 255) / 256, 256, 0, stream>>>(wq, wq_bf, n);
    cvt_f32_to_bf16<<<(n + 255) / 256, 256, 0, stream>>>(wo, wo_bf, n);
  }
  {
    int n = RANK * DIM;
    cvt_f32_to_bf16<<<(n + 255) / 256, 256, 0, stream>>>(wka, wka_bf, n);
    cvt_f32_to_bf16<<<(n + 255) / 256, 256, 0, stream>>>(wkb, wkb_bf, n);
  }

  // --- V head-mix (writes transposed bf16) ---
  {
    int n = 2 * NHEAD * HD * SEQ;
    vmix_kernel<<<(n + 255) / 256, 256, 0, stream>>>(xt, kron, vT_bf);
  }

  // --- Q projection: [4096,2048] = x @ wq^T ---
  gemm_bf16_wmma<0><<<(NTOK / 32) * (DIM / 64), 32, 0, stream>>>(
      x_bf, wq_bf, q_bf, NTOK, DIM, DIM);

  // --- K compress: [4096,512] = x @ wka^T (f32 out for RMSNorm) ---
  gemm_bf16_wmma<1><<<(NTOK / 32) * (RANK / 64), 32, 0, stream>>>(
      x_bf, wka_bf, kc_f, NTOK, RANK, DIM);

  // --- RMSNorm -> bf16 ---
  rmsnorm_kernel<<<(NTOK * 32 + 127) / 128, 128, 0, stream>>>(kc_f, knw, kc_bf, NTOK);

  // --- K expand: [4096,2048] = kc @ wkb^T ---
  gemm_bf16_wmma<0><<<(NTOK / 32) * (DIM / 64), 32, 0, stream>>>(
      kc_bf, wkb_bf, k_bf, NTOK, DIM, RANK);

  // --- flash attention (causal + ALiBi) ---
  attn_kernel<<<2 * NHEAD * (SEQ / 16), 32, 0, stream>>>(q_bf, k_bf, vT_bf, a_bf);

  // --- output projection: [4096,2048] f32 = attn @ wo^T ---
  gemm_bf16_wmma<1><<<(NTOK / 32) * (DIM / 64), 32, 0, stream>>>(
      a_bf, wo_bf, (float*)d_out, NTOK, DIM, DIM);
}
